// MyModel_42339787604606
// MI455X (gfx1250) — matrix operand order repaired and checked by reading, not compiled, not run on hardware
//
#include <hip/hip_runtime.h>
#include <hip/hip_bf16.h>
#include <stdint.h>

// ---------------------------------------------------------------------------
// Types matching gfx1250 WMMA / TDM builtin signatures
// ---------------------------------------------------------------------------
typedef __attribute__((ext_vector_type(16))) __bf16 v16bf;
typedef __attribute__((ext_vector_type(8)))  float  v8f;
typedef __attribute__((ext_vector_type(4)))  unsigned int v4u;
typedef __attribute__((ext_vector_type(8)))  int v8i_;
typedef __attribute__((ext_vector_type(4)))  int v4i_;

union FragBF { uint4 q[2]; v16bf v; };

#if defined(__has_builtin)
#if __has_builtin(__builtin_amdgcn_tensor_load_to_lds) && \
    __has_builtin(__builtin_amdgcn_s_wait_tensorcnt)
#define USE_TDM 1
#endif
#endif
#ifndef USE_TDM
#define USE_TDM 0
#endif

#define BATCH 16
#define CIN   512
#define NPIX  (BATCH * 32 * 32)   // 16384 output pixels
#define KTOT  (CIN * 9)           // 4608 reduction length (implicit GEMM)
#define BLK_M 128
#define BLK_N 128
#define BLK_K 32

// round-to-nearest-even f32 -> bf16
__device__ __forceinline__ unsigned short f2bf(float f) {
  unsigned u = __float_as_uint(f);
  unsigned r = (u + 0x7FFFu + ((u >> 16) & 1u)) >> 16;
  return (unsigned short)r;
}

__global__ void f32_to_bf16_kernel(const float* __restrict__ in,
                                   unsigned short* __restrict__ out, int n) {
  int i = blockIdx.x * blockDim.x + threadIdx.x;
  if (i < n) out[i] = f2bf(in[i]);
}

// ---------------------------------------------------------------------------
// 3x3 SAME conv as implicit GEMM with V_WMMA_F32_16X16X32_BF16.
//   D[oc, p] = sum_{k} W[oc, k] * Im2col[k, p],  k = ic*9 + ky*3 + kx
// Block tile: 128 (oc) x 128 (pix), K-step 32.  8 waves: 4 (M) x 2 (N),
// each wave computes 2x4 = 8 WMMA 16x16 tiles.
// A (weight) tile staged by the Tensor Data Mover (TENSOR_LOAD_TO_LDS):
//   tensor = [OC][KTOT] row-major, tile = 32 (K) x 128 (rows) -> lA[row*32+k].
//   TDM zero-fills rows >= OC (tensor_dim1 = OC), handling the cls layer pad.
// ---------------------------------------------------------------------------
__global__ __launch_bounds__(256)
void conv3x3_wmma_kernel(const unsigned short* __restrict__ in,   // bf16 [B][512][32][32]
                         const unsigned short* __restrict__ wgt,  // bf16 [OC][4608]
                         const float* __restrict__ bias,          // [OC]
                         unsigned short* __restrict__ outBf16,    // bf16 [B][OC][1024] or null
                         float* __restrict__ outF32,              // f32 same layout or null
                         int OC, int doRelu)
{
  __shared__ unsigned short lA[BLK_M * BLK_K]; // [row][k], row stride 32 (64B)
  __shared__ unsigned short lB[BLK_N * BLK_K]; // [pix][k], pixel-major for B frags

  const int tid    = threadIdx.x;
  const int lane   = tid & 31;
  const int wave   = tid >> 5;
  const int wm     = wave & 3;     // wave tile row (M)
  const int wn     = wave >> 2;    // wave tile col (N)
  const int laneM  = lane & 15;
  const int laneHi = lane >> 4;

  const int mBase = blockIdx.y * BLK_M;
  const int pBase = blockIdx.x * BLK_N;
  const int bImg  = pBase >> 10;   // one batch image per 128-pixel tile
  const int yx0   = pBase & 1023;

  v8f acc[2][4];
#pragma unroll
  for (int i = 0; i < 2; ++i)
#pragma unroll
    for (int j = 0; j < 4; ++j) {
      v8f z = {0.f, 0.f, 0.f, 0.f, 0.f, 0.f, 0.f, 0.f};
      acc[i][j] = z;
    }

#if USE_TDM
  // --- TDM descriptor (constant part), per CDNA5 ISA 8.3/8.4 ---
  // generic LDS pointer: low 32 bits are the LDS byte offset (aperture rule)
  const unsigned ldsA = (unsigned)(uintptr_t)&lA[0];
  v8i_ g1c;
  g1c[0] = (int)(1u << 16);                                   // data_size = 2 bytes
  g1c[1] = (int)(((unsigned)KTOT & 0xFFFFu) << 16);           // tensor_dim0[15:0]
  g1c[2] = (int)((((unsigned)KTOT >> 16) & 0xFFFFu) |
                 (((unsigned)OC & 0xFFFFu) << 16));           // dim0[31:16] | dim1[15:0]
  g1c[3] = (int)((((unsigned)OC >> 16) & 0xFFFFu) |
                 ((unsigned)BLK_K << 16));                    // dim1[31:16] | tile_dim0=32
  g1c[4] = BLK_M;                                             // tile_dim1=128, tile_dim2=0
  g1c[5] = KTOT;                                              // tensor_dim0_stride[31:0]
  g1c[6] = 0;                                                 // stride0[47:32] | stride1[15:0]
  g1c[7] = 0;
  const v4i_ gz4 = {0, 0, 0, 0};
  const v8i_ gz8 = {0, 0, 0, 0, 0, 0, 0, 0};
#else
  // A loader: 256 threads, each moves 16 bf16 (two b128) of a 128x32 tile
  const int aRow  = tid >> 1;
  const int aHalf = tid & 1;
#endif
  // B loader: 32 k-rows x 8 threads, each gathers 16 pixels (im2col)
  const int bKK = tid >> 3;
  const int bJ0 = (tid & 7) * 16;

  for (int k0 = 0; k0 < KTOT; k0 += BLK_K) {
#if USE_TDM
    // ---- stage A tile via Tensor Data Mover (wave 0 issues the DMA) ----
    if (wave == 0) {
      unsigned long long gaddr =
          (unsigned long long)(uintptr_t)(wgt + (size_t)mBase * KTOT + k0);
      v4u g0;
      g0.x = 1u;                                              // count=1 (valid, user mode)
      g0.y = ldsA;                                            // lds_addr
      g0.z = (unsigned)(gaddr & 0xFFFFFFFFull);               // global_addr[31:0]
      g0.w = (unsigned)((gaddr >> 32) & 0x1FFFFFFull) |
             (2u << 30);                                      // global_addr[56:32] | type=2
      __builtin_amdgcn_tensor_load_to_lds(g0, g1c, gz4, gz4, gz8, 0);
    }
#else
    // ---- stage A tile (synchronous fallback) ----
    {
      int grow = mBase + aRow;
      uint4* dst = (uint4*)&lA[aRow * BLK_K + aHalf * 16];
      if (grow < OC) {
        const uint4* src = (const uint4*)&wgt[(size_t)grow * KTOT + k0 + aHalf * 16];
        dst[0] = src[0];
        dst[1] = src[1];
      } else {
        uint4 z = make_uint4(0u, 0u, 0u, 0u);
        dst[0] = z; dst[1] = z;
      }
    }
#endif
    // ---- stage B tile (im2col gather with zero pad) ----
    {
      int k  = k0 + bKK;
      int ic = k / 9;
      int r  = k - ic * 9;
      int r3 = r / 3;
      int dy = r3 - 1;
      int dx = r - r3 * 3 - 1;
      const unsigned short* base = in + (((size_t)bImg * CIN + ic) << 10);
#pragma unroll
      for (int e = 0; e < 16; ++e) {
        int j  = bJ0 + e;
        int yx = yx0 + j;
        int y  = (yx >> 5) + dy;
        int x  = (yx & 31) + dx;
        unsigned short v = 0;
        if ((unsigned)y < 32u && (unsigned)x < 32u)
          v = base[(y << 5) + x];
        lB[j * BLK_K + bKK] = v;
      }
    }
    // hint next weight slab toward the caches (global_prefetch_b8)
    if (k0 + BLK_K < KTOT)
      __builtin_prefetch(&wgt[(size_t)(mBase + tid) * KTOT + k0 + BLK_K], 0, 1);

#if USE_TDM
    if (wave == 0)
      __builtin_amdgcn_s_wait_tensorcnt(0);   // TDM data landed in LDS
#endif
    __syncthreads();

    // ---- load fragments per ISA wave32 striping ----
    FragBF aF[2], bF[4];
#pragma unroll
    for (int mt = 0; mt < 2; ++mt) {
      int row = wm * 32 + mt * 16 + laneM;
      // lane<16: K 0..7 & 16..23 ; lane>=16: K 8..15 & 24..31
      const uint4* pa = (const uint4*)&lA[row * BLK_K + laneHi * 8];
      aF[mt].q[0] = pa[0];
      aF[mt].q[1] = pa[2];
    }
#pragma unroll
    for (int nt = 0; nt < 4; ++nt) {
      int pix = wn * 64 + nt * 16 + laneM;
      // lane<16: K 0..7 & 16..23 ; lane>=16: K 8..15 & 24..31
      const uint4* pb = (const uint4*)&lB[pix * BLK_K + laneHi * 8];
      bF[nt].q[0] = pb[0];
      bF[nt].q[1] = pb[2];
    }
#pragma unroll
    for (int mt = 0; mt < 2; ++mt)
#pragma unroll
      for (int nt = 0; nt < 4; ++nt)
        acc[mt][nt] = __builtin_amdgcn_wmma_f32_16x16x32_bf16(
            false, aF[mt].v, false, bF[nt].v, (short)0, acc[mt][nt], false, false);

    __syncthreads();
  }

  // ---- epilogue: bias (+ReLU), store bf16 and/or f32 ----
#pragma unroll
  for (int mt = 0; mt < 2; ++mt) {
#pragma unroll
    for (int nt = 0; nt < 4; ++nt) {
      int n  = pBase + wn * 64 + nt * 16 + laneM;
      int yx = n & 1023;
#pragma unroll
      for (int r8 = 0; r8 < 8; ++r8) {
        int oc = mBase + wm * 32 + mt * 16 + laneHi * 8 + r8;
        if (oc < OC) {
          float v = acc[mt][nt][r8] + bias[oc];
          if (doRelu) v = fmaxf(v, 0.0f);
          int ofs = ((bImg * OC + oc) << 10) + yx;
          if (outBf16) outBf16[ofs] = f2bf(v);
          if (outF32)  outF32[ofs]  = v;
        }
      }
    }
  }
}

// ---------------------------------------------------------------------------
// Tail kernels (tiny fraction of the FLOPs)
// ---------------------------------------------------------------------------
__global__ void reorder_logits_kernel(const float* __restrict__ clsOut,
                                      float* __restrict__ out, int n) {
  int o = blockIdx.x * blockDim.x + threadIdx.x;
  if (o >= n) return;
  int c  = o % 18;          // a*2 + k
  int p  = o / 18;          // b*1024 + yx
  int b  = p >> 10;
  int yx = p & 1023;
  out[o] = clsOut[((b * 18 + c) << 10) + yx];
}

__global__ __launch_bounds__(256)
void select_local_kernel(const float* __restrict__ logits,   // (16,9216,2)
                         const float* __restrict__ anchors,  // (9216,4)
                         const int* __restrict__ imgShape,
                         const float* __restrict__ features, // f32 (16,512,32,32)
                         float* __restrict__ local)          // (16,512)
{
  __shared__ float sv[256];
  __shared__ int   si[256];
  __shared__ int   sij[2];
  int b = blockIdx.x, tid = threadIdx.x;
  float best = -3.0e38f; int bidx = 0x7fffffff;
  for (int i = tid; i < 9216; i += 256) {
    // softmax p_fg is monotone in (l1 - l0): argmax of the diff
    float d = logits[(b * 9216 + i) * 2 + 1] - logits[(b * 9216 + i) * 2 + 0];
    if (d > best || (d == best && i < bidx)) { best = d; bidx = i; }
  }
  sv[tid] = best; si[tid] = bidx;
  __syncthreads();
  for (int s = 128; s > 0; s >>= 1) {
    if (tid < s) {
      float ov = sv[tid + s]; int oi = si[tid + s];
      if (ov > sv[tid] || (ov == sv[tid] && oi < si[tid])) { sv[tid] = ov; si[tid] = oi; }
    }
    __syncthreads();
  }
  if (tid == 0) {
    int idx = si[0];
    float decim = (float)(imgShape[0] / 32);
    float cx = (anchors[idx * 4 + 2] - anchors[idx * 4 + 0]) / decim;
    float cy = (anchors[idx * 4 + 3] - anchors[idx * 4 + 1]) / decim;
    sij[0] = (int)fminf(fmaxf(rintf(cy), 0.f), 31.f);
    sij[1] = (int)fminf(fmaxf(rintf(cx), 0.f), 31.f);
  }
  __syncthreads();
  int ii = sij[0], jj = sij[1];
  for (int c = tid; c < 512; c += 256)
    local[b * 512 + c] = features[((b * 512 + c) << 10) + (ii << 5) + jj];
}

__global__ void bn_batch_kernel(const float* __restrict__ x, const float* __restrict__ g,
                                const float* __restrict__ be, float* __restrict__ y) {
  int c = blockIdx.x * blockDim.x + threadIdx.x;
  if (c >= 512) return;
  float s = 0.f, s2 = 0.f;
  for (int i = 0; i < 16; ++i) { float v = x[i * 512 + c]; s += v; s2 += v * v; }
  float m   = s / 16.f;
  float inv = rsqrtf(s2 / 16.f - m * m + 1e-5f);
  for (int i = 0; i < 16; ++i)
    y[i * 512 + c] = g[c] * (x[i * 512 + c] - m) * inv + be[c];
}

__global__ void mlp_kernel(const float* __restrict__ in, const float* __restrict__ W,
                           float* __restrict__ out, int K, int NOUT, int doRelu) {
  int t = blockIdx.x * blockDim.x + threadIdx.x;
  if (t >= 16 * NOUT) return;
  int i = t / NOUT, n = t % NOUT;
  float s = 0.f;
  for (int k = 0; k < K; ++k) s += in[i * K + k] * W[n * K + k];
  if (doRelu) s = fmaxf(s, 0.f);
  out[t] = s;
}

__global__ void scale_kernel(const float* __restrict__ local, const float* __restrict__ w,
                             const float* __restrict__ bsc, const float* __restrict__ g,
                             const float* __restrict__ be, float* __restrict__ scale) {
  __shared__ float lin[16];
  __shared__ float mv[2];
  int i = threadIdx.x;
  if (i < 16) {
    float s = 0.f;
    for (int k = 0; k < 512; ++k) s += local[i * 512 + k] * w[k];
    lin[i] = s + bsc[0];
  }
  __syncthreads();
  if (i == 0) {
    float s = 0.f, s2 = 0.f;
    for (int j = 0; j < 16; ++j) { s += lin[j]; s2 += lin[j] * lin[j]; }
    float m = s / 16.f;
    mv[0] = m; mv[1] = rsqrtf(s2 / 16.f - m * m + 1e-5f);
  }
  __syncthreads();
  if (i < 16)
    scale[i] = expf(g[0] * (lin[i] - mv[0]) * mv[1] + be[0]);
}

__global__ void l2norm_kernel(const float* __restrict__ emb, float* __restrict__ xn) {
  __shared__ float red[128];
  int i = blockIdx.x, t = threadIdx.x; // 128 threads
  float e = emb[i * 128 + t];
  red[t] = e * e;
  __syncthreads();
  for (int s = 64; s > 0; s >>= 1) { if (t < s) red[t] += red[t + s]; __syncthreads(); }
  float norm = fmaxf(sqrtf(red[0]), 1e-12f);
  xn[i * 128 + t] = e / norm;
}

__global__ void cosine_kernel(const float* __restrict__ xn, const float* __restrict__ cw,
                              const float* __restrict__ scale, float* __restrict__ cosOut) {
  int t = blockIdx.x * blockDim.x + threadIdx.x;
  if (t >= 16 * 100) return;
  int i = t / 100, j = t % 100;
  float dot = 0.f, n2 = 0.f;
  for (int c = 0; c < 128; ++c) {
    float wv = cw[j * 128 + c];
    dot += xn[i * 128 + c] * wv;
    n2  += wv * wv;
  }
  cosOut[i * 100 + j] = scale[i] * dot / fmaxf(sqrtf(n2), 1e-12f);
}

// ---------------------------------------------------------------------------
extern "C" void kernel_launch(void* const* d_in, const int* in_sizes, int n_in,
                              void* d_out, int out_size, void* d_ws, size_t ws_size,
                              hipStream_t stream) {
  (void)in_sizes; (void)n_in; (void)out_size; (void)ws_size;

  const float* features = (const float*)d_in[0];
  const float* anchors  = (const float*)d_in[1];
  const int*   imgShape = (const int*)d_in[2];
  const float* conv_w[4] = {(const float*)d_in[3], (const float*)d_in[5],
                            (const float*)d_in[7], (const float*)d_in[9]};
  const float* conv_b[4] = {(const float*)d_in[4], (const float*)d_in[6],
                            (const float*)d_in[8], (const float*)d_in[10]};
  const float* cls_w = (const float*)d_in[11];
  const float* cls_b = (const float*)d_in[12];
  const float* bn_g  = (const float*)d_in[13];
  const float* bn_b  = (const float*)d_in[14];
  const float* pw1   = (const float*)d_in[15];
  const float* pw2   = (const float*)d_in[16];
  const float* pw3   = (const float*)d_in[17];
  const float* cw    = (const float*)d_in[18];
  const float* fsw   = (const float*)d_in[19];
  const float* fsb   = (const float*)d_in[20];
  const float* bsg   = (const float*)d_in[21];
  const float* bsb   = (const float*)d_in[22];

  char* ws = (char*)d_ws;
  size_t off = 0;
  auto alloc = [&](size_t bytes) {
    void* p = ws + off;
    off += (bytes + 255) & ~(size_t)255;
    return p;
  };
  const size_t ACT_N = 16ull * 512 * 32 * 32;   // 8,388,608
  const size_t WCV_N = 512ull * 512 * 9;        // 2,359,296 per layer
  unsigned short* actA  = (unsigned short*)alloc(ACT_N * 2);
  unsigned short* actB  = (unsigned short*)alloc(ACT_N * 2);
  unsigned short* wconv = (unsigned short*)alloc(4 * WCV_N * 2);
  unsigned short* wcls  = (unsigned short*)alloc(18ull * 4608 * 2);
  float* clsOut = (float*)alloc(294912ull * 4);
  float* local  = (float*)alloc(16 * 512 * 4);
  float* bnloc  = (float*)alloc(16 * 512 * 4);
  float* h1     = (float*)alloc(16 * 512 * 4);
  float* h2     = (float*)alloc(16 * 512 * 4);
  float* emb    = (float*)alloc(16 * 128 * 4);
  float* scl    = (float*)alloc(16 * 4);

  float* outLogits = (float*)d_out;          // (16,9216,2)
  float* outCos    = outLogits + 294912;     // (16,100)
  float* outXn     = outCos + 1600;          // (16,128)

  // f32 -> bf16 staging (feed the matrix pipe)
  f32_to_bf16_kernel<<<(int)((ACT_N + 255) / 256), 256, 0, stream>>>(features, actA, (int)ACT_N);
  for (int l = 0; l < 4; ++l)
    f32_to_bf16_kernel<<<(int)((WCV_N + 255) / 256), 256, 0, stream>>>(
        conv_w[l], wconv + (size_t)l * WCV_N, (int)WCV_N);
  f32_to_bf16_kernel<<<(18 * 4608 + 255) / 256, 256, 0, stream>>>(cls_w, wcls, 18 * 4608);

  // 4 x (conv + bias + ReLU) via WMMA implicit GEMM, ping-pong activations
  dim3 convGrid(NPIX / BLK_N, 4);  // 128 pixel tiles x 4 oc tiles
  conv3x3_wmma_kernel<<<convGrid, 256, 0, stream>>>(actA, wconv + 0 * WCV_N, conv_b[0], actB, nullptr, 512, 1);
  conv3x3_wmma_kernel<<<convGrid, 256, 0, stream>>>(actB, wconv + 1 * WCV_N, conv_b[1], actA, nullptr, 512, 1);
  conv3x3_wmma_kernel<<<convGrid, 256, 0, stream>>>(actA, wconv + 2 * WCV_N, conv_b[2], actB, nullptr, 512, 1);
  conv3x3_wmma_kernel<<<convGrid, 256, 0, stream>>>(actB, wconv + 3 * WCV_N, conv_b[3], actA, nullptr, 512, 1);

  // cls head (OC=18, no relu, f32 out); TDM zero-fills padded rows 18..127
  dim3 clsGrid(NPIX / BLK_N, 1);
  conv3x3_wmma_kernel<<<clsGrid, 256, 0, stream>>>(actA, wcls, cls_b, nullptr, clsOut, 18, 0);

  reorder_logits_kernel<<<(294912 + 255) / 256, 256, 0, stream>>>(clsOut, outLogits, 294912);
  select_local_kernel<<<16, 256, 0, stream>>>(outLogits, anchors, imgShape, features, local);
  bn_batch_kernel<<<2, 256, 0, stream>>>(local, bn_g, bn_b, bnloc);
  mlp_kernel<<<(16 * 512 + 255) / 256, 256, 0, stream>>>(bnloc, pw1, h1, 512, 512, 1);
  mlp_kernel<<<(16 * 512 + 255) / 256, 256, 0, stream>>>(h1, pw2, h2, 512, 512, 1);
  mlp_kernel<<<(16 * 128 + 255) / 256, 256, 0, stream>>>(h2, pw3, emb, 512, 128, 0);
  scale_kernel<<<1, 32, 0, stream>>>(local, fsw, fsb, bsg, bsb, scl);
  l2norm_kernel<<<16, 128, 0, stream>>>(emb, outXn);
  cosine_kernel<<<(1600 + 255) / 256, 256, 0, stream>>>(outXn, cw, scl, outCos);
}
